// CrossSS2D_4114578669725
// MI455X (gfx1250) — matrix operand order repaired and checked by reading, not compiled, not run on hardware
//
#include <hip/hip_runtime.h>
#include <hip/hip_bf16.h>
#include <math.h>

// ---------------- problem constants ----------------
static constexpr int Bb  = 8;
static constexpr int HH  = 48;
static constexpr int WW  = 48;
static constexpr int LL  = HH * WW;          // 2304
static constexpr int Dm  = 96;               // d_model
static constexpr int Din = 192;              // d_inner
static constexpr int Ns  = 16;               // n_state
static constexpr int Rr  = 6;                // dt_rank
static constexpr int Kk  = 4;                // directions
static constexpr int CXP = Rr + 2 * Ns;      // 38 rows of x_dbl
static constexpr int CXPAD = 48;             // padded to 3 x 16 WMMA tiles
static constexpr int BL  = Bb * LL;          // 18432

typedef __attribute__((ext_vector_type(16))) __bf16 v16bf;
typedef __attribute__((ext_vector_type(8)))  float  v8f;
typedef __attribute__((ext_vector_type(4)))  unsigned int u32x4;
typedef __attribute__((ext_vector_type(4)))  int    i32x4;
typedef __attribute__((ext_vector_type(8)))  int    i32x8;

// native RNE conversion -> v_cvt_pk_bf16_f32 on gfx1250
static __device__ __forceinline__ __bf16 f2bf(float f) { return (__bf16)f; }

// canonical (row-major) index for scan position l of direction k
static __device__ __forceinline__ int dirmap(int k, int l) {
    switch (k) {
        case 0: return l;
        case 1: return (l % WW) * WW + (l / WW);           // transpose (H==W)
        case 2: return LL - 1 - l;                          // flipped row-major
        default: { int t = LL - 1 - l; return (t % WW) * WW + (t / WW); }
    }
}

// ==========================================================================
// GEMM 1/2: C[M x N] = A[M x 96] * W[N x 96]^T   (bf16 WMMA, K = 96 = 3*32)
// one wave per 16x16 output tile
// ==========================================================================
__global__ void k_gemm_proj(const float* __restrict__ A,
                            const float* __restrict__ W,
                            float* __restrict__ C, int M, int N) {
    const int ntiles = N >> 4;
    const int mt = blockIdx.x / ntiles, nt = blockIdx.x % ntiles;
    const int m0 = mt << 4, n0 = nt << 4;
    const int lane = threadIdx.x & 31;
    const int half = lane >> 4, r = lane & 15;

    v8f acc = {};
    for (int k0 = 0; k0 < 96; k0 += 32) {
        v16bf av, bv;
        // A fragment (16x32): lanes 0-15: row r, K {0..7,16..23}; 16-31: row r, K {8..15,24..31}
        const float* arow = A + (size_t)(m0 + r) * 96 + k0 + (half ? 8 : 0);
#pragma unroll
        for (int j = 0; j < 8; ++j) { av[j] = f2bf(arow[j]); av[j + 8] = f2bf(arow[16 + j]); }
        // B fragment (32x16): lanes 0-15: col r, K {0..7,16..23}; 16-31: col r, K {8..15,24..31}
        const float* wrow = W + (size_t)(n0 + r) * 96 + k0 + (half ? 8 : 0);
#pragma unroll
        for (int j = 0; j < 8; ++j) { bv[j] = f2bf(wrow[j]); bv[j + 8] = f2bf(wrow[16 + j]); }
        acc = __builtin_amdgcn_wmma_f32_16x16x32_bf16(false, av, false, bv,
                                                      (short)0, acc, false, false);
    }
    const int n = n0 + r, mb = m0 + (half ? 8 : 0);
#pragma unroll
    for (int i = 0; i < 8; ++i) C[(size_t)(mb + i) * N + n] = acc[i];
}

// ==========================================================================
// depthwise 3x3 conv (SAME, pad 1) + bias + SiLU, channel-last in & out
// ==========================================================================
__global__ void k_conv_silu(const float* __restrict__ in, int stride, int coff,
                            const float* __restrict__ cw, const float* __restrict__ cb,
                            float* __restrict__ out) {
    const int idx = blockIdx.x * blockDim.x + threadIdx.x;
    if (idx >= Bb * LL * Din) return;
    const int d  = idx % Din;
    const int bl = idx / Din;
    const int l0 = bl % LL, b = bl / LL;
    const int h = l0 / WW, w = l0 % WW;
    float acc = cb[d];
#pragma unroll
    for (int dh = -1; dh <= 1; ++dh) {
#pragma unroll
        for (int dw = -1; dw <= 1; ++dw) {
            const int hh = h + dh, ww = w + dw;
            if (hh >= 0 && hh < HH && ww >= 0 && ww < WW)
                acc += in[((size_t)b * LL + hh * WW + ww) * stride + coff + d] *
                       cw[d * 9 + (dh + 1) * 3 + (dw + 1)];
        }
    }
    acc = acc / (1.f + __expf(-acc));  // SiLU
    out[(size_t)bl * Din + d] = acc;
}

// ==========================================================================
// x_proj GEMM: per (b,k)  x_dbl[48 x L] = x_proj_w[k] (38x192, zero-pad to 48)
//                                        * kvs_k (192 x L, direction-gathered)
// ==========================================================================
__global__ void k_gemm_xproj(const float* __restrict__ xw,   // (K,38,192)
                             const float* __restrict__ kvc,  // (B,L,192) canonical
                             float* __restrict__ xdbl) {     // (B,K,48,L)
    int id = blockIdx.x;
    const int nt = id % (LL / 16); id /= (LL / 16);
    const int mt = id % 3;         id /= 3;
    const int b = id / Kk, k = id % Kk;
    const int m0 = mt << 4, n0 = nt << 4;
    const int lane = threadIdx.x & 31;
    const int half = lane >> 4, r = lane & 15;
    const int c  = m0 + r;                 // A row (c-dim, valid < 38)
    const int l  = n0 + r;                 // B col (scan position)
    const size_t brow = ((size_t)b * LL + dirmap(k, l)) * Din;

    v8f acc = {};
    for (int k0 = 0; k0 < Din; k0 += 32) {
        v16bf av, bv;
        if (c < CXP) {
            const float* arow = xw + ((size_t)k * CXP + c) * Din + k0 + (half ? 8 : 0);
#pragma unroll
            for (int j = 0; j < 8; ++j) { av[j] = f2bf(arow[j]); av[j + 8] = f2bf(arow[16 + j]); }
        } else {
#pragma unroll
            for (int j = 0; j < 16; ++j) av[j] = f2bf(0.f);
        }
        const float* bsrc = kvc + brow + k0 + (half ? 8 : 0);
#pragma unroll
        for (int j = 0; j < 8; ++j) { bv[j] = f2bf(bsrc[j]); bv[j + 8] = f2bf(bsrc[16 + j]); }
        acc = __builtin_amdgcn_wmma_f32_16x16x32_bf16(false, av, false, bv,
                                                      (short)0, acc, false, false);
    }
    float* dst = xdbl + ((size_t)(b * Kk + k) * CXPAD) * LL;
    const int cb = m0 + (half ? 8 : 0);
#pragma unroll
    for (int i = 0; i < 8; ++i) dst[(size_t)(cb + i) * LL + (n0 + r)] = acc[i];
}

// ==========================================================================
// selective scan: one lane per (b,k,d), 16 states in registers,
// 192-thread block per (b,k).  Per-16-step chunk, the Tensor Data Mover DMAs
// the 38x16 x_dbl tile (row stride L) into LDS rows (tensor_load_to_lds),
// synchronized with s_wait_tensorcnt + workgroup barriers.
// dt projection (rank 6) + softplus fused. ys stored channel-last (b,k,l,d).
// ==========================================================================
__global__ void __launch_bounds__(Din) k_scan(const float* __restrict__ xdbl,
                                              const float* __restrict__ qc,
                                              const float* __restrict__ A_logs,
                                              const float* __restrict__ dt_w,
                                              const float* __restrict__ dt_b,
                                              float* __restrict__ ys) {
    const int bk = blockIdx.x;
    const int b = bk / Kk, k = bk % Kk;
    const int d = threadIdx.x;

    float a[Ns], h[Ns], dw[Rr];
#pragma unroll
    for (int n = 0; n < Ns; ++n) {
        a[n] = -__expf(A_logs[((size_t)(k * Din + d)) * Ns + n]);
        h[n] = 0.f;
    }
#pragma unroll
    for (int r = 0; r < Rr; ++r) dw[r] = dt_w[(size_t)(k * Din + d) * Rr + r];
    const float db = dt_b[k * Din + d];

    __shared__ float xb[CXP * 16];           // 38 x 16 chunk of x_dbl (TDM dest)
    const float* xsrc = xdbl + (size_t)bk * CXPAD * LL;
    const float* xq   = qc + (size_t)b * LL * Din;
    float* yout = ys + (size_t)bk * LL * Din;

    // TDM descriptor invariants (2D tensor: 48 rows x L cols, fp32)
    const unsigned lds_base = (unsigned)(size_t)&xb[0];

    for (int lc = 0; lc < LL; lc += 16) {
        __syncthreads();                      // previous chunk fully consumed
        if (threadIdx.x < 32) {
            const unsigned long long ga = (unsigned long long)(size_t)(xsrc + lc);
            u32x4 g0;
            g0[0] = 1u;                                   // count=1, user mode
            g0[1] = lds_base;                             // LDS byte address
            g0[2] = (unsigned)(ga & 0xFFFFFFFFull);       // global addr [31:0]
            g0[3] = (unsigned)((ga >> 32) & 0x01FFFFFFull) | (2u << 30); // [56:32] | type=2
            i32x8 g1;
            g1[0] = 0x00020000;                           // data_size=2 (4B), no multicast
            g1[1] = (int)(((unsigned)LL & 0xFFFFu) << 16);// tensor_dim0[15:0]
            g1[2] = (int)(((unsigned)LL >> 16) | ((unsigned)CXPAD << 16)); // dim0 hi | dim1 lo
            g1[3] = (int)(16u << 16);                     // dim1 hi=0 | tile_dim0=16
            g1[4] = (int)CXP;                             // tile_dim1=38, tile_dim2=0
            g1[5] = (int)LL;                              // tensor_dim0_stride lo
            g1[6] = 0;                                    // stride hi | dim1_stride lo
            g1[7] = 0;                                    // dim1_stride hi
            i32x4 gz = {0, 0, 0, 0};
#if __clang_major__ >= 23
            i32x8 gz8 = {0, 0, 0, 0, 0, 0, 0, 0};
            __builtin_amdgcn_tensor_load_to_lds(g0, g1, gz, gz, gz8, 0);
#else
            __builtin_amdgcn_tensor_load_to_lds(g0, g1, gz, gz, 0);
#endif
            __builtin_amdgcn_s_wait_tensorcnt(0);
        }
        __syncthreads();                      // chunk visible to all waves
#pragma unroll 4
        for (int t = 0; t < 16; ++t) {
            const int l = lc + t;
            float dtr = db;
#pragma unroll
            for (int r = 0; r < Rr; ++r) dtr += dw[r] * xb[r * 16 + t];
            const float dtv = (dtr > 20.f) ? dtr : log1pf(__expf(dtr));
            const float xv  = xq[(size_t)dirmap(k, l) * Din + d];
            const float dx  = dtv * xv;
            float y = 0.f;
#pragma unroll
            for (int n = 0; n < Ns; ++n) {
                const float dA = __expf(dtv * a[n]);
                h[n] = h[n] * dA + dx * xb[(Rr + n) * 16 + t];
                y += h[n] * xb[(Rr + Ns + n) * 16 + t];
            }
            yout[(size_t)l * Din + d] = y;
        }
    }
}

// ==========================================================================
// direction merge (gathers with flip/transpose maps) + D*x skip + LayerNorm
// + gate add (z) ; writes bf16 rows for the output GEMM
// ==========================================================================
__global__ void __launch_bounds__(Din) k_merge_ln(const float* __restrict__ ys,
                                                  const float* __restrict__ qc,
                                                  const float* __restrict__ Ds,
                                                  const float* __restrict__ qp,
                                                  const float* __restrict__ lnw,
                                                  const float* __restrict__ lnb,
                                                  __bf16* __restrict__ ybf) {
    const int d  = threadIdx.x;
    const int bl = blockIdx.x;
    const int l0 = bl % LL, b = bl / LL;
    const int l1 = (l0 % WW) * WW + (l0 / WW);
    const size_t base = (size_t)b * Kk;

    float v = ys[((base + 0) * LL + l0)           * Din + d]
            + ys[((base + 1) * LL + l1)           * Din + d]
            + ys[((base + 2) * LL + (LL - 1 - l0)) * Din + d]
            + ys[((base + 3) * LL + (LL - 1 - l1)) * Din + d];
    const float dssum = Ds[d] + Ds[Din + d] + Ds[2 * Din + d] + Ds[3 * Din + d];
    v += qc[(size_t)bl * Din + d] * dssum;

    float s = v, s2 = v * v;
#pragma unroll
    for (int o = 16; o; o >>= 1) { s += __shfl_xor(s, o, 32); s2 += __shfl_xor(s2, o, 32); }
    __shared__ float ws1[6], ws2[6], st[2];
    const int wid = d >> 5, lane = d & 31;
    if (lane == 0) { ws1[wid] = s; ws2[wid] = s2; }
    __syncthreads();
    if (d == 0) {
        float t1 = 0.f, t2 = 0.f;
        for (int i = 0; i < 6; ++i) { t1 += ws1[i]; t2 += ws2[i]; }
        const float mu = t1 / (float)Din;
        st[0] = mu;
        st[1] = rsqrtf(t2 / (float)Din - mu * mu + 1e-5f);
    }
    __syncthreads();
    const float z  = qp[((size_t)b * LL + l0) * (2 * Din) + Din + d];
    const float yn = (v - st[0]) * st[1] * lnw[d] + lnb[d] + z;
    ybf[(size_t)bl * Din + d] = f2bf(yn);
}

// ==========================================================================
// out-proj GEMM: (BL x 192) bf16 * out_proj_w(96 x 192)^T, scatter-store to
// (B, 96, H, W)
// ==========================================================================
__global__ void k_gemm_out(const __bf16* __restrict__ A,
                           const float* __restrict__ W,
                           float* __restrict__ out) {
    const int ntiles = Dm >> 4;  // 6
    const int mt = blockIdx.x / ntiles, nt = blockIdx.x % ntiles;
    const int m0 = mt << 4, n0 = nt << 4;
    const int lane = threadIdx.x & 31;
    const int half = lane >> 4, r = lane & 15;

    v8f acc = {};
    for (int k0 = 0; k0 < Din; k0 += 32) {
        v16bf av, bv;
        const __bf16* arow = A + (size_t)(m0 + r) * Din + k0 + (half ? 8 : 0);
#pragma unroll
        for (int j = 0; j < 8; ++j) { av[j] = arow[j]; av[j + 8] = arow[16 + j]; }
        const float* wrow = W + (size_t)(n0 + r) * Din + k0 + (half ? 8 : 0);
#pragma unroll
        for (int j = 0; j < 8; ++j) { bv[j] = f2bf(wrow[j]); bv[j + 8] = f2bf(wrow[16 + j]); }
        acc = __builtin_amdgcn_wmma_f32_16x16x32_bf16(false, av, false, bv,
                                                      (short)0, acc, false, false);
    }
    const int n = n0 + r, mb = m0 + (half ? 8 : 0);
#pragma unroll
    for (int i = 0; i < 8; ++i) {
        const int m = mb + i;
        const int b = m / LL, l = m % LL;
        out[((size_t)(b * Dm + n) * HH + l / WW) * WW + (l % WW)] = acc[i];
    }
}

// ==========================================================================
extern "C" void kernel_launch(void* const* d_in, const int* in_sizes, int n_in,
                              void* d_out, int out_size, void* d_ws, size_t ws_size,
                              hipStream_t stream) {
    const float* q_x   = (const float*)d_in[0];
    const float* kv_x  = (const float*)d_in[1];
    const float* w1    = (const float*)d_in[2];   // (384, 96)
    const float* w2    = (const float*)d_in[3];   // (192, 96)
    const float* cw    = (const float*)d_in[4];   // (192,1,3,3)
    const float* cb    = (const float*)d_in[5];
    const float* xpw   = (const float*)d_in[6];   // (4,38,192)
    const float* dtw   = (const float*)d_in[7];   // (4,192,6)
    const float* dtb   = (const float*)d_in[8];   // (4,192)
    const float* Alog  = (const float*)d_in[9];   // (4,192,16)
    const float* Ds    = (const float*)d_in[10];  // (4,192)
    const float* lnw   = (const float*)d_in[11];
    const float* lnb   = (const float*)d_in[12];
    const float* wout  = (const float*)d_in[13];  // (96,192)
    float* out = (float*)d_out;

    // workspace carve-up (floats)
    float* ws = (float*)d_ws;
    size_t o = 0;
    float* qp   = ws + o; o += (size_t)BL * (2 * Din);          // (B,L,384)
    float* kvp  = ws + o; o += (size_t)BL * Din;                 // (B,L,192)
    float* qc   = ws + o; o += (size_t)BL * Din;                 // conv(q),  (B,L,192)
    float* kvc  = ws + o; o += (size_t)BL * Din;                 // conv(kv), (B,L,192)
    float* xdbl = ws + o; o += (size_t)Bb * Kk * CXPAD * LL;     // (B,K,48,L)
    float* ys   = ws + o; o += (size_t)Bb * Kk * LL * Din;       // (B,K,L,192)
    __bf16* ybf = (__bf16*)(ws + o);                             // (B,L,192) bf16

    // 1) input projections (WMMA bf16)
    k_gemm_proj<<<(BL / 16) * (2 * Din / 16), 32, 0, stream>>>(q_x, w1, qp, BL, 2 * Din);
    k_gemm_proj<<<(BL / 16) * (Din / 16),     32, 0, stream>>>(kv_x, w2, kvp, BL, Din);

    // 2) depthwise conv + SiLU (channel-last)
    const int cvN = Bb * LL * Din;
    k_conv_silu<<<cvN / 256, 256, 0, stream>>>(qp, 2 * Din, 0, cw, cb, qc);
    k_conv_silu<<<cvN / 256, 256, 0, stream>>>(kvp, Din, 0, cw, cb, kvc);

    // 3) x_proj per direction (WMMA bf16, direction-gathered B operand)
    k_gemm_xproj<<<Bb * Kk * 3 * (LL / 16), 32, 0, stream>>>(xpw, kvc, xdbl);

    // 4) selective scan: 32 blocks x 192 threads (one lane per channel), TDM-staged
    k_scan<<<Bb * Kk, Din, 0, stream>>>(xdbl, qc, Alog, dtw, dtb, ys);

    // 5) direction merge + skip + LayerNorm + gate
    k_merge_ln<<<BL, Din, 0, stream>>>(ys, qc, Ds, qp, lnw, lnb, ybf);

    // 6) out projection (WMMA bf16) with (B,96,H,W) scatter store
    k_gemm_out<<<(BL / 16) * (Dm / 16), 32, 0, stream>>>(ybf, wout, out);
}
